// MultiHeadAttention_23424751633194
// MI455X (gfx1250) — matrix operand order repaired and checked by reading, not compiled, not run on hardware
//
#include <hip/hip_runtime.h>

// ---------------------------------------------------------------------------
// CDNA5 (gfx1250) fused MHA: bf16 WMMA projections + fused attention.
//   EMBED=1024, H=16, d=64, rows M = B*S = 65536, attention groups of 128 rows.
// ---------------------------------------------------------------------------

typedef __attribute__((ext_vector_type(16))) __bf16         v16bf;
typedef __attribute__((ext_vector_type(8)))  float          v8f;
typedef __attribute__((ext_vector_type(8)))  unsigned short v8us;
typedef __attribute__((ext_vector_type(4)))  int            v4i;

union Frag {            // 32 bytes: one WMMA A/B operand (16 bf16)
    v16bf bf;
    v8us  h[2];
};

__device__ __forceinline__ unsigned short f2bf(float f) {
    unsigned u = __builtin_bit_cast(unsigned, f);
    u += 0x7FFFu + ((u >> 16) & 1u);          // round-to-nearest-even
    return (unsigned short)(u >> 16);
}
__device__ __forceinline__ unsigned pk2(float a, float b) {
    return (unsigned)f2bf(a) | ((unsigned)f2bf(b) << 16);
}

__device__ __forceinline__ v8f wmma_bf16(const Frag& a, const Frag& b, v8f c) {
    return __builtin_amdgcn_wmma_f32_16x16x32_bf16(
        false, a.bf, false, b.bf, (short)0, c, false, false);
}

// ---- optional gfx1250 async global->LDS path (guarded; VGPR copy fallback) --
#if __has_builtin(__builtin_amdgcn_global_load_async_to_lds_b128) && \
    __has_builtin(__builtin_amdgcn_s_wait_asynccnt)
#define HAVE_ASYNC_LDS 1
typedef __attribute__((address_space(1))) v4i gv4i;   // global (as1) v4i
typedef __attribute__((address_space(3))) v4i lv4i;   // LDS    (as3) v4i
__device__ __forceinline__ void async_cp16(const unsigned short* g, unsigned short* l) {
    // flat->addrspace via integer casts: global flat VA == as(1) VA;
    // low 32 bits of a flat LDS address are the LDS byte offset (aperture trunc).
    __builtin_amdgcn_global_load_async_to_lds_b128(
        (gv4i*)(unsigned long long)g,
        (lv4i*)(unsigned)(unsigned long long)l,
        0, 0);
}
#else
#define HAVE_ASYNC_LDS 0
#endif

// ---------------------------------------------------------------------------
// Kernel A: Y[m,n] = sum_k X[m,k] * W[n,k] + bias[n]   (i.e. X @ W.T + b)
// Output written as bf16 (consumed by the attention kernel).
// Block: 256 threads = 8 waves; tile 128(M) x 128(N); K-step 64 (2 WMMA K).
// ---------------------------------------------------------------------------
#define KBLK 64
#define LDA  (KBLK + 8)     // 72 ushorts: 144B row stride -> aligned, bank-spread

__global__ __launch_bounds__(256)
void qkv_gemm(const float* __restrict__ X, const float* __restrict__ W,
              const float* __restrict__ bias, unsigned short* __restrict__ Y)
{
    __shared__ unsigned short sA[128 * LDA];
    __shared__ unsigned short sB[128 * LDA];

    const int tid   = threadIdx.x;
    const int lane  = tid & 31;
    const int wid   = tid >> 5;
    const int waveM = wid & 3;       // 4 waves along M (32 rows each)
    const int waveN = wid >> 2;      // 2 waves along N (64 cols each)
    const int m0    = blockIdx.y * 128;
    const int n0    = blockIdx.x * 128;

    const int row   = lane & 15;
    const int hi    = lane >> 4;

    v8f acc[2][4] = {};

    const int srow  = tid >> 1;          // staging: row 0..127
    const int skseg = (tid & 1) * 32;    // K half 0 / 32

    const float* gx = X + (size_t)(m0 + srow) * 1024 + skseg;
    const float* gw = W + (size_t)(n0 + srow) * 1024 + skseg;

    for (int kb = 0; kb < 1024; kb += KBLK) {
        // prefetch next K-panel (X streams from HBM; W mostly L2-resident)
        if (kb + KBLK < 1024) {
            __builtin_prefetch(gx + kb + KBLK, 0, 1);
            __builtin_prefetch(gw + kb + KBLK, 0, 1);
        }
        // ---- stage X tile (fp32 -> bf16) ----
        {
            const float4* px = (const float4*)(gx + kb);
            float4 x0 = px[0], x1 = px[1], x2 = px[2], x3 = px[3];
            float4 x4 = px[4], x5 = px[5], x6 = px[6], x7 = px[7];
            uint4* dst = (uint4*)&sA[srow * LDA + skseg];
            dst[0] = make_uint4(pk2(x0.x,x0.y), pk2(x0.z,x0.w), pk2(x1.x,x1.y), pk2(x1.z,x1.w));
            dst[1] = make_uint4(pk2(x2.x,x2.y), pk2(x2.z,x2.w), pk2(x3.x,x3.y), pk2(x3.z,x3.w));
            dst[2] = make_uint4(pk2(x4.x,x4.y), pk2(x4.z,x4.w), pk2(x5.x,x5.y), pk2(x5.z,x5.w));
            dst[3] = make_uint4(pk2(x6.x,x6.y), pk2(x6.z,x6.w), pk2(x7.x,x7.y), pk2(x7.z,x7.w));
        }
        // ---- stage W tile (rows of W == columns of B operand) ----
        {
            const float4* pw = (const float4*)(gw + kb);
            float4 x0 = pw[0], x1 = pw[1], x2 = pw[2], x3 = pw[3];
            float4 x4 = pw[4], x5 = pw[5], x6 = pw[6], x7 = pw[7];
            uint4* dst = (uint4*)&sB[srow * LDA + skseg];
            dst[0] = make_uint4(pk2(x0.x,x0.y), pk2(x0.z,x0.w), pk2(x1.x,x1.y), pk2(x1.z,x1.w));
            dst[1] = make_uint4(pk2(x2.x,x2.y), pk2(x2.z,x2.w), pk2(x3.x,x3.y), pk2(x3.z,x3.w));
            dst[2] = make_uint4(pk2(x4.x,x4.y), pk2(x4.z,x4.w), pk2(x5.x,x5.y), pk2(x5.z,x5.w));
            dst[3] = make_uint4(pk2(x6.x,x6.y), pk2(x6.z,x6.w), pk2(x7.x,x7.y), pk2(x7.z,x7.w));
        }
        __syncthreads();

        // ---- fragments (ISA 16-bit A 16x32 / B 32x16 layouts), 2 K-steps ----
        for (int ks = 0; ks < 2; ++ks) {
            Frag af[2], bfr[4];
            for (int mt = 0; mt < 2; ++mt) {
                const unsigned short* p = &sA[(waveM*32 + mt*16 + row) * LDA + ks*32 + hi*8];
                af[mt].h[0] = *(const v8us*)(p);
                af[mt].h[1] = *(const v8us*)(p + 16);
            }
            for (int nt = 0; nt < 4; ++nt) {
                const unsigned short* p = &sB[(waveN*64 + nt*16 + row) * LDA + ks*32 + hi*8];
                bfr[nt].h[0] = *(const v8us*)(p);
                bfr[nt].h[1] = *(const v8us*)(p + 16);
            }
            for (int mt = 0; mt < 2; ++mt)
                for (int nt = 0; nt < 4; ++nt)
                    acc[mt][nt] = wmma_bf16(af[mt], bfr[nt], acc[mt][nt]);
        }
        __syncthreads();
    }

    // ---- epilogue: +bias, bf16 store ----
    for (int mt = 0; mt < 2; ++mt)
        for (int nt = 0; nt < 4; ++nt) {
            const int colg = n0 + waveN*64 + nt*16 + row;
            const float bv = bias[colg];
            const int rb   = m0 + waveM*32 + mt*16 + hi*8;
            for (int r = 0; r < 8; ++r)
                Y[(size_t)(rb + r) * 1024 + colg] = f2bf(acc[mt][nt][r] + bv);
        }
}

// ---------------------------------------------------------------------------
// Kernel B: fused attention per (group n, head h).
//   Q,K,V tiles 128x64 bf16; S = QK^T/8 (128x128); softmax; O = P V (128x64).
// Block: 128 threads = 4 waves; wave w owns output rows [32w, 32w+32).
// ---------------------------------------------------------------------------
#define SQK 72     // Q/K LDS row stride (ushorts), 144B: aligned + bank-spread
#define SPV 136    // P / V^T LDS row stride, 272B

__global__ __launch_bounds__(128)
void attn(const unsigned short* __restrict__ Qb, const unsigned short* __restrict__ Kb,
          const unsigned short* __restrict__ Vb, float* __restrict__ Out)
{
    __shared__ unsigned short sQ [128 * SQK];
    __shared__ unsigned short sK [128 * SQK];
    __shared__ unsigned short sVT[ 64 * SPV];
    __shared__ unsigned short sP [128 * SPV];

    const int h    = blockIdx.x;
    const int n    = blockIdx.y;
    const int tid  = threadIdx.x;
    const int lane = tid & 31;
    const int w    = tid >> 5;
    const int row  = lane & 15;
    const int hi   = lane >> 4;

    const size_t baseRow = (size_t)n * 128;
    const size_t colOff  = (size_t)h * 64;

    // ---- stage: Q,K rows via async global->LDS (pure copy); V scattered to V^T
    {
        const unsigned short* q = Qb + (baseRow + tid) * 1024 + colOff;
        const unsigned short* k = Kb + (baseRow + tid) * 1024 + colOff;
        unsigned short* dq = &sQ[tid * SQK];
        unsigned short* dk = &sK[tid * SQK];
#if HAVE_ASYNC_LDS
        for (int i = 0; i < 8; ++i) {
            async_cp16(q + i*8, dq + i*8);
            async_cp16(k + i*8, dk + i*8);
        }
#else
        for (int i = 0; i < 8; ++i) {
            ((v8us*)dq)[i] = ((const v8us*)q)[i];
            ((v8us*)dk)[i] = ((const v8us*)k)[i];
        }
#endif
        const v8us* v = (const v8us*)(Vb + (baseRow + tid) * 1024 + colOff);
        for (int c = 0; c < 8; ++c) {
            v8us vv = v[c];
            for (int j = 0; j < 8; ++j) sVT[(c*8 + j) * SPV + tid] = vv[j];
        }
    }
#if HAVE_ASYNC_LDS
    __builtin_amdgcn_s_wait_asynccnt(0);
#endif
    __syncthreads();

    // ---- S = Q K^T : M=128 (rows 32w..), N=128, K=64 (2 WMMA K-steps)
    v8f acc[2][8] = {};
    for (int ks = 0; ks < 2; ++ks) {
        Frag af[2];
        for (int mt = 0; mt < 2; ++mt) {
            const unsigned short* p = &sQ[(w*32 + mt*16 + row) * SQK + ks*32 + hi*8];
            af[mt].h[0] = *(const v8us*)(p);
            af[mt].h[1] = *(const v8us*)(p + 16);
        }
        for (int nt = 0; nt < 8; ++nt) {
            Frag bfr;
            const unsigned short* p = &sK[(nt*16 + row) * SQK + ks*32 + hi*8];
            bfr.h[0] = *(const v8us*)(p);
            bfr.h[1] = *(const v8us*)(p + 16);
            for (int mt = 0; mt < 2; ++mt)
                acc[mt][nt] = wmma_bf16(af[mt], bfr, acc[mt][nt]);
        }
    }

    // ---- softmax (rows span lanes 0..15 of each half; shfl_xor <=8 stays in half)
    float rinv[2][8];
    for (int mt = 0; mt < 2; ++mt)
        for (int r = 0; r < 8; ++r) {
            float m = -1e30f;
            for (int nt = 0; nt < 8; ++nt) {
                acc[mt][nt][r] *= 0.125f;              // 1/sqrt(d), d=64
                m = fmaxf(m, acc[mt][nt][r]);
            }
            for (int off = 8; off; off >>= 1) m = fmaxf(m, __shfl_xor(m, off, 32));
            float s = 0.f;
            for (int nt = 0; nt < 8; ++nt) {
                float e = __expf(acc[mt][nt][r] - m);
                acc[mt][nt][r] = e;
                s += e;
            }
            for (int off = 8; off; off >>= 1) s += __shfl_xor(s, off, 32);
            rinv[mt][r] = __builtin_amdgcn_rcpf(s);    // normalize after P@V
        }

    // ---- write unnormalized P (bf16) to LDS for A-operand relayout
    for (int mt = 0; mt < 2; ++mt)
        for (int nt = 0; nt < 8; ++nt) {
            const int M = w*32 + mt*16 + hi*8;
            const int N = nt*16 + row;
            for (int r = 0; r < 8; ++r)
                sP[(M + r) * SPV + N] = f2bf(acc[mt][nt][r]);
        }
    __syncthreads();

    // ---- O = P V : M=128, K=128 (4 steps), N=64 ; B operand from V^T rows
    v8f o[2][4] = {};
    for (int ks = 0; ks < 4; ++ks) {
        Frag af[2];
        for (int mt = 0; mt < 2; ++mt) {
            const unsigned short* p = &sP[(w*32 + mt*16 + row) * SPV + ks*32 + hi*8];
            af[mt].h[0] = *(const v8us*)(p);
            af[mt].h[1] = *(const v8us*)(p + 16);
        }
        for (int nt = 0; nt < 4; ++nt) {
            Frag bfr;
            const unsigned short* p = &sVT[(nt*16 + row) * SPV + ks*32 + hi*8];
            bfr.h[0] = *(const v8us*)(p);
            bfr.h[1] = *(const v8us*)(p + 16);
            for (int mt = 0; mt < 2; ++mt)
                o[mt][nt] = wmma_bf16(af[mt], bfr, o[mt][nt]);
        }
    }

    // ---- normalize rows and store fp32 output (same flat layout as input)
    for (int mt = 0; mt < 2; ++mt)
        for (int nt = 0; nt < 4; ++nt) {
            const int M = w*32 + mt*16 + hi*8;
            const int N = nt*16 + row;
            for (int r = 0; r < 8; ++r)
                Out[(baseRow + M + r) * 1024 + colOff + N] = o[mt][nt][r] * rinv[mt][r];
        }
}

// ---------------------------------------------------------------------------
extern "C" void kernel_launch(void* const* d_in, const int* in_sizes, int n_in,
                              void* d_out, int out_size, void* d_ws, size_t ws_size,
                              hipStream_t stream) {
    const float* q  = (const float*)d_in[0];
    const float* k  = (const float*)d_in[1];
    const float* v  = (const float*)d_in[2];
    const float* Wq = (const float*)d_in[3];
    const float* bq = (const float*)d_in[4];
    const float* Wk = (const float*)d_in[5];
    const float* bk = (const float*)d_in[6];
    const float* Wv = (const float*)d_in[7];
    const float* bv = (const float*)d_in[8];
    float* out = (float*)d_out;

    const size_t MN = (size_t)65536 * 1024;     // rows * EMBED
    unsigned short* qb = (unsigned short*)d_ws; // 3 x 128 MB bf16 scratch
    unsigned short* kb = qb + MN;
    unsigned short* vb = kb + MN;

    dim3 gG(1024 / 128, 65536 / 128);           // (8, 512)
    qkv_gemm<<<gG, 256, 0, stream>>>(q, Wq, bq, qb);
    qkv_gemm<<<gG, 256, 0, stream>>>(k, Wk, bk, kb);
    qkv_gemm<<<gG, 256, 0, stream>>>(v, Wv, bv, vb);

    dim3 gA(16, 512);                           // (head, group)
    attn<<<gA, 128, 0, stream>>>(qb, kb, vb, out);
}
